// ActQuantizer_23158463660205
// MI455X (gfx1250) — matrix operand order repaired and checked by reading, not compiled, not run on hardware
//
#include <hip/hip_runtime.h>

// ActQuantizer: group-wise Hadamard rotation (group=128) + symmetric int8 quant.
//
// MI455X reasoning: ~516 MiB of traffic -> ~22us floor at 23.3 TB/s. The
// Hadamard-as-GEMM is 256 FLOP/elem = 17.2 GFLOP: negligible on f16 WMMA.
// H entries are +-1 (exact in f16); accumulate in f32 via
// v_wmma_f32_16x16x32_f16. H128 factors so every 32x16 B-slice is one of just
// 4 register-resident matrices built from H16 = (-1)^popc(i&j):
//   B(kb,jb) = s1 * [[H16],[(-1)^(jb&1) H16]],  s1 = (-1)^popc(kb & (jb>>1))
// so B never touches memory.

typedef __attribute__((ext_vector_type(16))) _Float16 v16h;
typedef __attribute__((ext_vector_type(8)))  float    v8f;

#define RSQRT_G 0.08838834764831845f   // 1/sqrt(128)

__global__ __launch_bounds__(256) void hadq_int8_kernel(
    const float* __restrict__ x,
    float* __restrict__ scale_out,
    float* __restrict__ zp_out,
    float* __restrict__ xint_out,
    int ntiles) {
  const int lane    = threadIdx.x & 31;
  const int wave    = threadIdx.x >> 5;
  const int tile    = blockIdx.x * 8 + wave;     // one wave = 16 groups of 128
  if (tile >= ntiles) return;                    // wave-uniform exit

  const int lcol    = lane & 15;                 // 0..15
  const int halfsel = lane >> 4;                 // 0 or 1

  // ---------------- Build the 4 B matrices in registers ----------------
  // B layout (f16, 32x16): lanes 0-15 column n=lane, rows K=0..15 in 8 VGPRs
  // (2 halfs/VGPR); lanes 16-31 column n=lane-16, rows K=16..31.
  v16h Bp, Bm, Bpn, Bmn;   // [[H],[H]], [[H],[-H]], and negations
#pragma unroll
  for (int k = 0; k < 16; ++k) {
    const int par = __popc(k & lcol) & 1;
    const float s = par ? -1.0f : 1.0f;
    const float sm = halfsel ? -s : s;           // second 16 rows of Bm flip
    Bp[k]  = (_Float16)s;
    Bpn[k] = (_Float16)(-s);
    Bm[k]  = (_Float16)sm;
    Bmn[k] = (_Float16)(-sm);
  }

  // ---------------- Load A tile (16x128 f32), cvt to f16 ----------------
  // A layout per k-block (16x32 f16): lanes 0-15: M=lane, K = 0..15;
  // lanes 16-31: M=lane-16, K = 16..31.  Fold 1/sqrt(128) here.
  const float* xb = x + (size_t)tile * 2048 + (size_t)lcol * 128 + halfsel * 16;
  v16h A[4];
#pragma unroll
  for (int kb = 0; kb < 4; ++kb) {
    float t[16];
    *(float4*)&t[0]  = *(const float4*)(xb + kb * 32 + 0);
    *(float4*)&t[4]  = *(const float4*)(xb + kb * 32 + 4);
    *(float4*)&t[8]  = *(const float4*)(xb + kb * 32 + 8);
    *(float4*)&t[12] = *(const float4*)(xb + kb * 32 + 12);
#pragma unroll
    for (int e = 0; e < 16; ++e) A[kb][e] = (_Float16)(t[e] * RSQRT_G);
  }

  // ---------------- 32 WMMAs: Y[16x128] = A x H-slices ----------------
  v8f C[8];
#pragma unroll
  for (int jb = 0; jb < 8; ++jb) {
    v8f acc = {};
#pragma unroll
    for (int kb = 0; kb < 4; ++kb) {
      const int s1 = __popc(kb & (jb >> 1)) & 1;       // compile-time
      const v16h Bsel = (jb & 1) ? (s1 ? Bmn : Bm) : (s1 ? Bpn : Bp);
      acc = __builtin_amdgcn_wmma_f32_16x16x32_f16(
          /*neg_a=*/false, A[kb], /*neg_b=*/false, Bsel,
          /*c_mod=*/(short)0, acc, /*reuse_a=*/false, /*reuse_b=*/false);
    }
    C[jb] = acc;
  }

  // ---------------- Per-row absmax (row = v + 8*halfsel) ----------------
  // C layout: VGPR v, lanes 0-15 -> (M=v, N=lane); lanes 16-31 -> (M=v+8).
  float scl[8], inv[8];
#pragma unroll
  for (int v = 0; v < 8; ++v) {
    float m = 0.0f;
#pragma unroll
    for (int jb = 0; jb < 8; ++jb) m = fmaxf(m, fabsf(C[jb][v]));
    // butterfly max across the 16 lanes of this half (xor 8,4,2,1)
#pragma unroll
    for (int off = 8; off >= 1; off >>= 1)
      m = fmaxf(m, __shfl_xor(m, off, 32));
    const float s = fmaxf(m * (1.0f / 127.0f), 1e-8f);
    scl[v] = s;
    inv[v] = 1.0f / s;
  }

  // ---------------- Quantize + store x_int ----------------
  float* ob = xint_out + (size_t)tile * 2048 + (size_t)halfsel * 8 * 128 + lcol;
#pragma unroll
  for (int v = 0; v < 8; ++v) {
#pragma unroll
    for (int jb = 0; jb < 8; ++jb) {
      float q = rintf(C[jb][v] * inv[v]);           // RTE, matches jnp.round
      q = fminf(fmaxf(q, -128.0f), 127.0f);         // v_med3-able clamp
      ob[(size_t)v * 128 + jb * 16] = q;
    }
  }

  // ---------------- Store scale / zero_point (16 rows per tile) ----------
  float sv = scl[0];
#pragma unroll
  for (int v = 1; v < 8; ++v) sv = (lcol == v) ? scl[v] : sv;
  if (lcol < 8) {
    const int r = tile * 16 + halfsel * 8 + lcol;
    scale_out[r] = sv;
    zp_out[r]    = 0.0f;
  }
}

extern "C" void kernel_launch(void* const* d_in, const int* in_sizes, int n_in,
                              void* d_out, int out_size, void* d_ws, size_t ws_size,
                              hipStream_t stream) {
  (void)n_in; (void)out_size; (void)d_ws; (void)ws_size;
  const float* x = (const float*)d_in[0];
  const long long n = (long long)in_sizes[0];     // B*S*D = 67108864
  const int R = (int)(n / 128);                   // number of groups = 524288
  const int ntiles = (R + 15) / 16;               // 32768 (exact for this shape)

  float* scale = (float*)d_out;                   // [R]
  float* zp    = scale + R;                       // [R]
  float* xint  = zp + R;                          // [n]

  const int waves_per_block = 8;                  // 256 threads, wave32
  const int blocks = (ntiles + waves_per_block - 1) / waves_per_block;
  hadq_int8_kernel<<<blocks, 256, 0, stream>>>(x, scale, zp, xint, ntiles);
}
